// frcnn_net_24266565222974
// MI455X (gfx1250) — matrix operand order repaired and checked by reading, not compiled, not run on hardware
//
#include <hip/hip_runtime.h>
#include <hip/hip_bf16.h>

typedef unsigned short u16;
typedef __attribute__((ext_vector_type(16))) __bf16 v16bf;
typedef __attribute__((ext_vector_type(8)))  float  v8f;

#define NBOX   500
#define NROW   1000          // B*N real rows
#define MPAD   1024          // feat rows padded (zero-filled) so A staging needs no guard
#define CCH    512
#define HFDIM  25
#define KDIM   25088         // C*7*7
#define FCN    4096
#define BM 128
#define BN 128
#define BK 32
#define LDA 40               // bf16 elems per LDS row (80B rows, 16B aligned)
#define LDW 40
#define NITER (KDIM / BK)    // 784

__device__ __forceinline__ u16 f2bf(float f) {
    union { __bf16 h; u16 u; } v;
    v.h = (__bf16)f;         // hardware RNE f32->bf16 cvt
    return v.u;
}

// ---------------------------------------------------------------------------
// Kernel 1: ROI pool (adaptive avg pool 7x7 on int-cast box crops) -> bf16 feat
// One block per padded row; rows >= 1000 are zero-filled (GEMM M padding).
// ---------------------------------------------------------------------------
__global__ __launch_bounds__(256) void roi_pool_kernel(
    const float* __restrict__ fm, const float* __restrict__ boxes,
    u16* __restrict__ feat)
{
    const int row = blockIdx.x;          // b*500 + n  (or pad row)
    const int t   = threadIdx.x;

    if (row >= NROW) {                   // zero pad rows 1000..1023
        uint4 z = {};
        uint4* dst = (uint4*)(feat + (long)row * KDIM);
        for (int k = t; k < KDIM / 8; k += 256) dst[k] = z;
        return;
    }

    const int b = row / NBOX;
    __shared__ int rs[7], re_[7], cs[7], ce_[7];
    const float* bx = boxes + (long)row * 4;
    if (t < 7) {
        int r0 = (int)(bx[0] / 32.0f);
        int c0 = (int)(bx[1] / 32.0f);
        int r1 = (int)(bx[2] / 32.0f);
        int c1 = (int)(bx[3] / 32.0f);
        int sr = r1 - r0, sc = c1 - c0;
        rs[t]  = r0 + (t * sr) / 7;
        re_[t] = r0 + ((t + 1) * sr + 6) / 7;
        cs[t]  = c0 + (t * sc) / 7;
        ce_[t] = c0 + ((t + 1) * sc + 6) / 7;
    }
    __syncthreads();

    const float* fmb = fm + (long)b * CCH * HFDIM * HFDIM;
    u16* frow = feat + (long)row * KDIM;

    for (int k = t; k < KDIM; k += 256) {
        int c   = k / 49;
        int bin = k - c * 49;
        int i   = bin / 7;
        int j   = bin - i * 7;
        const float* plane = fmb + (long)c * (HFDIM * HFDIM);
        int r0 = rs[i], r1 = re_[i], c0 = cs[j], c1 = ce_[j];
        float s = 0.0f;
        for (int r = r0; r < r1; ++r)
            for (int cc = c0; cc < c1; ++cc)
                s += plane[r * HFDIM + cc];
        float area = (float)((r1 - r0) * (c1 - c0));
        frow[k] = f2bf(s / area);
    }
}

// ---------------------------------------------------------------------------
// Kernel 2: fc = feat[1024x25088](bf16) x fc_w[4096x25088]^T (+fc_b), f32 acc
// via v_wmma_f32_16x16x32_bf16. 128x128x32 tiles, 256 thr = 8 wave32s, wave
// sub-tile 32x64. Double-buffered LDS (1 barrier / K-step). A tile staged via
// async direct-to-LDS copies (ASYNCcnt); W tile converted f32->bf16 in regs.
// ---------------------------------------------------------------------------
__global__ __launch_bounds__(256) void gemm_fc_kernel(
    const u16*  __restrict__ feat,
    const float* __restrict__ fc_w,
    const float* __restrict__ fc_b,
    float* __restrict__ fc)
{
    __shared__ __align__(16) u16 aT[2][BM * LDA];
    __shared__ __align__(16) u16 wT[2][BN * LDW];

    const int tid  = threadIdx.x;
    const int lane = tid & 31;
    const int wave = tid >> 5;
    const int wm   = wave & 3;          // 0..3 -> 32-row slice
    const int wn   = wave >> 2;         // 0..1 -> 64-col slice
    const int l16  = lane & 15;
    const int hi   = lane >> 4;

    const int n0 = blockIdx.x * BN;
    const int m0 = blockIdx.y * BM;

    union Acc  { v8f  v; float f[8]; };
    union Frag { v16bf v; uint4 q[2]; };

    Acc acc[2][4];
    v8f zero = {};
    #pragma unroll
    for (int mf = 0; mf < 2; ++mf)
        #pragma unroll
        for (int nf = 0; nf < 4; ++nf) acc[mf][nf].v = zero;

    // staging: 2 threads per tile row, 16 elements (32B) each
    const int arow  = tid >> 1;
    const int ahalf = (tid & 1) * 16;                 // bf16 offset
    const u16* aSrc = feat + (long)(m0 + arow) * KDIM + ahalf;  // padded, no guard
    unsigned aLds[2];
    aLds[0] = (unsigned)(unsigned long long)&aT[0][arow * LDA + ahalf];
    aLds[1] = (unsigned)(unsigned long long)&aT[1][arow * LDA + ahalf];

    const int wrow  = tid >> 1;
    const int whalf = (tid & 1) * 16;                 // f32 offset
    const float* wSrc = fc_w + (long)(n0 + wrow) * KDIM + whalf;

    // --- prologue: stage tile 0 into buffer 0 ---
    {
        unsigned long long g0 = (unsigned long long)aSrc;
        asm volatile("global_load_async_to_lds_b128 %0, %1, off"
                     :: "v"(aLds[0]), "v"(g0) : "memory");
        asm volatile("global_load_async_to_lds_b128 %0, %1, off offset:16"
                     :: "v"(aLds[0]), "v"(g0) : "memory");
        float4 f0 = *(const float4*)(wSrc);
        float4 f1 = *(const float4*)(wSrc + 4);
        float4 f2 = *(const float4*)(wSrc + 8);
        float4 f3 = *(const float4*)(wSrc + 12);
        u16* wd = &wT[0][wrow * LDW + whalf];
        wd[0]=f2bf(f0.x); wd[1]=f2bf(f0.y); wd[2]=f2bf(f0.z); wd[3]=f2bf(f0.w);
        wd[4]=f2bf(f1.x); wd[5]=f2bf(f1.y); wd[6]=f2bf(f1.z); wd[7]=f2bf(f1.w);
        wd[8]=f2bf(f2.x); wd[9]=f2bf(f2.y); wd[10]=f2bf(f2.z); wd[11]=f2bf(f2.w);
        wd[12]=f2bf(f3.x); wd[13]=f2bf(f3.y); wd[14]=f2bf(f3.z); wd[15]=f2bf(f3.w);
        asm volatile("s_wait_asynccnt 0x0" ::: "memory");
    }
    __syncthreads();

    for (int it = 0; it < NITER; ++it) {
        const int  buf  = it & 1;
        const int  k0n  = (it + 1) * BK;
        const bool more = (it + 1 < NITER);

        // --- issue next tile's loads first (overlap with WMMA) ---
        float4 f0 = {}, f1 = {}, f2 = {}, f3 = {};
        if (more) {
            unsigned long long g0 = (unsigned long long)(aSrc + k0n);
            asm volatile("global_load_async_to_lds_b128 %0, %1, off"
                         :: "v"(aLds[buf ^ 1]), "v"(g0) : "memory");
            asm volatile("global_load_async_to_lds_b128 %0, %1, off offset:16"
                         :: "v"(aLds[buf ^ 1]), "v"(g0) : "memory");
            f0 = *(const float4*)(wSrc + k0n);
            f1 = *(const float4*)(wSrc + k0n + 4);
            f2 = *(const float4*)(wSrc + k0n + 8);
            f3 = *(const float4*)(wSrc + k0n + 12);
        }

        // --- fragments from current buffer (ISA 7.12.2 lane layouts) ---
        Frag fa[2], fb[4];
        #pragma unroll
        for (int mf = 0; mf < 2; ++mf) {
            const u16* ar = &aT[buf][(wm * 32 + mf * 16 + l16) * LDA];
            fa[mf].q[0] = *(const uint4*)(ar + hi * 8);        // K 0-7 / 8-15
            fa[mf].q[1] = *(const uint4*)(ar + 16 + hi * 8);   // K 16-23 / 24-31
        }
        #pragma unroll
        for (int nf = 0; nf < 4; ++nf) {
            const u16* br = &wT[buf][(wn * 64 + nf * 16 + l16) * LDW];
            fb[nf].q[0] = *(const uint4*)(br + hi * 8);        // K 0-7 / 8-15
            fb[nf].q[1] = *(const uint4*)(br + 16 + hi * 8);   // K 16-23 / 24-31
        }

        #pragma unroll
        for (int mf = 0; mf < 2; ++mf)
            #pragma unroll
            for (int nf = 0; nf < 4; ++nf)
                acc[mf][nf].v = __builtin_amdgcn_wmma_f32_16x16x32_bf16(
                    false, fa[mf].v, false, fb[nf].v,
                    (short)0, acc[mf][nf].v, false, false);

        // --- convert + store next W tile into the other buffer ---
        if (more) {
            u16* wd = &wT[buf ^ 1][wrow * LDW + whalf];
            wd[0]=f2bf(f0.x); wd[1]=f2bf(f0.y); wd[2]=f2bf(f0.z); wd[3]=f2bf(f0.w);
            wd[4]=f2bf(f1.x); wd[5]=f2bf(f1.y); wd[6]=f2bf(f1.z); wd[7]=f2bf(f1.w);
            wd[8]=f2bf(f2.x); wd[9]=f2bf(f2.y); wd[10]=f2bf(f2.z); wd[11]=f2bf(f2.w);
            wd[12]=f2bf(f3.x); wd[13]=f2bf(f3.y); wd[14]=f2bf(f3.z); wd[15]=f2bf(f3.w);
        }
        asm volatile("s_wait_asynccnt 0x0" ::: "memory");
        __syncthreads();
    }

    // --- writeback: C/D layout VGPR g -> M = g + 8*hi, N = l16 ---
    #pragma unroll
    for (int mf = 0; mf < 2; ++mf) {
        const int gmBase = m0 + wm * 32 + mf * 16 + hi * 8;
        #pragma unroll
        for (int nf = 0; nf < 4; ++nf) {
            const int gn = n0 + wn * 64 + nf * 16 + l16;
            const float bias = fc_b[gn];
            #pragma unroll
            for (int g = 0; g < 8; ++g) {
                const int gm = gmBase + g;
                if (gm < NROW)
                    fc[(long)gm * FCN + gn] = acc[mf][nf].f[g] + bias;
            }
        }
    }
}

// ---------------------------------------------------------------------------
// Kernel 3: heads. Per row: l_fc[80], s_fc[21] (+bias), argmax(s_fc),
// l_out = l_fc[mi : mi+4]. d_out: [0,4000)=l_out; [4000,25000)=s_fc.
// ---------------------------------------------------------------------------
__global__ __launch_bounds__(128) void heads_kernel(
    const float* __restrict__ fc,
    const float* __restrict__ l_w, const float* __restrict__ l_b,
    const float* __restrict__ s_w, const float* __restrict__ s_b,
    float* __restrict__ out)
{
    const int row = blockIdx.x;
    const int t   = threadIdx.x;

    __shared__ float rbuf[FCN];
    __shared__ float outs[101];
    __shared__ int   mi_sh;

    const float* fr = fc + (long)row * FCN;
    for (int k = t; k < FCN; k += 128) rbuf[k] = fr[k];
    __syncthreads();

    if (t < 101) {
        const float* w;
        float bias;
        if (t < 80) { w = l_w + (long)t * FCN;        bias = l_b[t]; }
        else        { w = s_w + (long)(t - 80) * FCN; bias = s_b[t - 80]; }
        float s = 0.0f;
        #pragma unroll 4
        for (int k = 0; k < FCN; ++k) s = fmaf(rbuf[k], w[k], s);
        outs[t] = s + bias;
    }
    __syncthreads();

    if (t == 0) {
        int mi = 0; float mv = outs[80];
        for (int j = 1; j < 21; ++j)
            if (outs[80 + j] > mv) { mv = outs[80 + j]; mi = j; }   // first max
        mi_sh = mi;
    }
    __syncthreads();

    if (t < 21) out[4000 + row * 21 + t] = outs[80 + t];
    if (t < 4)  out[row * 4 + t] = outs[mi_sh + t];
}

// ---------------------------------------------------------------------------
extern "C" void kernel_launch(void* const* d_in, const int* in_sizes, int n_in,
                              void* d_out, int out_size, void* d_ws, size_t ws_size,
                              hipStream_t stream) {
    const float* fm    = (const float*)d_in[0];
    const float* boxes = (const float*)d_in[1];
    const float* fc_w  = (const float*)d_in[2];
    const float* fc_b  = (const float*)d_in[3];
    const float* l_w   = (const float*)d_in[4];
    const float* l_b   = (const float*)d_in[5];
    const float* s_w   = (const float*)d_in[6];
    const float* s_b   = (const float*)d_in[7];
    float* out = (float*)d_out;

    // ws layout: [0, 51380224) feat bf16 (1024 x 25088 u16, rows >=1000 zero)
    //            [51380224, ...) fc f32 (1000 x 4096)
    u16*   feat = (u16*)d_ws;
    float* fc   = (float*)((char*)d_ws + (size_t)MPAD * KDIM * sizeof(u16));

    roi_pool_kernel<<<MPAD, 256, 0, stream>>>(fm, boxes, feat);

    dim3 g2(FCN / BN, MPAD / BM);   // 32 x 8
    gemm_fc_kernel<<<g2, 256, 0, stream>>>(feat, fc_w, fc_b, fc);

    heads_kernel<<<NROW, 128, 0, stream>>>(fc, l_w, l_b, s_w, s_b, out);
}
